// CriticSNN_62835371541037
// MI455X (gfx1250) — matrix operand order repaired and checked by reading, not compiled, not run on hardware
//
#include <hip/hip_runtime.h>
#include <hip/hip_bf16.h>
#include <math.h>

typedef __attribute__((ext_vector_type(16))) _Float16 v16h;
typedef __attribute__((ext_vector_type(8)))  _Float16 v8h;
typedef __attribute__((ext_vector_type(4)))  _Float16 v4h;
typedef __attribute__((ext_vector_type(8)))  float    v8f;
typedef __attribute__((ext_vector_type(4)))  float    v4f;

constexpr int BATCH  = 65536;
constexpr int H      = 128;
constexpr int SDIM   = 6;
constexpr int TSTEPS = 8;

// ---- dynamic LDS layout (bytes) ----
constexpr int OFF_WREC1 = 0;            // 128x128 f16
constexpr int OFF_WFC2  = 32768;        // 128x128 f16
constexpr int OFF_WREC2 = 65536;        // 128x128 f16
constexpr int OFF_WFC1  = 98304;        // 128x8 f32 (6 used per row)
constexpr int OFF_WMEAN = 102400;       // 128 f32
constexpr int OFF_WSTD  = 102912;       // 128 f32
constexpr int OFF_GRP   = 103424;       // per-row-group area
// per group: spk1[2] 8192 + spk2[2] 8192 + stbuf 512
constexpr int GRP_BYTES = 16896;
constexpr int NGROUPS   = 8;            // 512 threads = 16 waves = 8 pairs
constexpr int LDS_TOTAL = OFF_GRP + NGROUPS * GRP_BYTES;   // 238592 B < 320KB

__device__ __forceinline__ v16h vcat(v8h lo, v8h hi) {
  v16h r;
#pragma unroll
  for (int i = 0; i < 8; ++i) { r[i] = lo[i]; r[i + 8] = hi[i]; }
  return r;
}

// A operand (16x32 f16, MxK): spike buffer 16 rows x 128 halfs, row-major.
// Lanes 0-15: row=lane, a[0..7]=K(k0+0..7), a[8..15]=K(k0+16..23)
// Lanes 16-31: row=lane-16, a[0..7]=K(k0+8..15), a[8..15]=K(k0+24..31)
__device__ __forceinline__ v16h load_a(const _Float16* spk, int lane, int k0) {
  const int row = lane & 15;
  const int kh  = (lane >> 4) * 8;
  const _Float16* p = spk + row * H + k0 + kh;
  return vcat(*(const v8h*)p, *(const v8h*)(p + 16));
}

// B operand (32x16 f16, KxN): weight LDS row-major [128 out][128 in] f16,
// B(k,n) = W[n0+n][k]. Lane n<16 -> K=k0..k0+7,k0+16..23; lane>=16 -> K=k0+8..15,k0+24..31.
__device__ __forceinline__ v16h load_b(const _Float16* w, int lane, int n0, int k0) {
  const int n  = n0 + (lane & 15);
  const int kb = (lane >> 4) * 8;
  const _Float16* p = w + n * H + k0 + kb;
  return vcat(*(const v8h*)p, *(const v8h*)(p + 16));
}

__device__ __forceinline__ v8f wmma_f16(v16h a, v16h b, v8f c) {
  return __builtin_amdgcn_wmma_f32_16x16x32_f16(false, a, false, b, (short)0, c,
                                                false, false);
}

__global__ void __launch_bounds__(512)
snn_critic_kernel(const float* __restrict__ state,
                  const float* __restrict__ w_fc1,
                  const float* __restrict__ w_rec1,
                  const float* __restrict__ w_fc2,
                  const float* __restrict__ w_rec2,
                  const float* __restrict__ w_mean,
                  const float* __restrict__ w_std,
                  const float* __restrict__ alpha1,
                  const float* __restrict__ beta1,
                  const float* __restrict__ thr1p,
                  const float* __restrict__ alpha2,
                  const float* __restrict__ beta2,
                  const float* __restrict__ thr2p,
                  float* __restrict__ out) {
  extern __shared__ char smem[];
  _Float16* wrec1  = (_Float16*)(smem + OFF_WREC1);
  _Float16* wfc2s  = (_Float16*)(smem + OFF_WFC2);
  _Float16* wrec2  = (_Float16*)(smem + OFF_WREC2);
  float*    wfc1s  = (float*)(smem + OFF_WFC1);
  float*    wmeans = (float*)(smem + OFF_WMEAN);
  float*    wstds  = (float*)(smem + OFF_WSTD);

  const int tid  = threadIdx.x;
  const int lane = tid & 31;
  const int wv   = tid >> 5;       // 0..15
  const int grp  = wv >> 1;        // 0..7 : row group (16 batch rows)
  const int hlf  = wv & 1;         // 0/1  : N-half owned by this wave

  char* gbase = smem + OFF_GRP + grp * GRP_BYTES;
  _Float16* spk1d = (_Float16*)gbase;                 // 2 x (16x128) f16
  _Float16* spk2d = (_Float16*)(gbase + 8192);        // 2 x (16x128) f16
  float*    stbuf = (float*)(gbase + 16384);          // 16x8 f32

  // ---- stage weights to LDS (f32 -> f16) ----
  {
    const float* src[3] = {w_rec1, w_fc2, w_rec2};
    _Float16*    dst[3] = {wrec1, wfc2s, wrec2};
#pragma unroll
    for (int m = 0; m < 3; ++m) {
      const v4f* s4 = (const v4f*)src[m];
      v4h*       d4 = (v4h*)dst[m];
      for (int i = tid; i < (H * H) / 4; i += 512) {
        v4f v = s4[i];
        v4h hh;
        hh.x = (_Float16)v.x; hh.y = (_Float16)v.y;
        hh.z = (_Float16)v.z; hh.w = (_Float16)v.w;
        d4[i] = hh;
      }
    }
  }
  for (int i = tid; i < H * SDIM; i += 512) {
    int r = i / SDIM, c = i - r * SDIM;
    wfc1s[r * 8 + c] = w_fc1[i];
  }
  if (tid < H) { wmeans[tid] = w_mean[tid]; wstds[tid] = w_std[tid]; }

  // ---- stage this group's 16 state rows (one wave of the pair) ----
  const int rowbase = blockIdx.x * 128 + grp * 16;
  if (hlf == 0 && lane < 16) {
    const float* s = state + (size_t)(rowbase + lane) * SDIM;
#pragma unroll
    for (int k = 0; k < SDIM; ++k) stbuf[lane * 8 + k] = s[k];
  }

  // ---- zero read-buffers (buffer 0) of spike LDS: 64 threads per group ----
  {
    const int gt = hlf * 32 + lane;        // 0..63 within group
    v8h z;
#pragma unroll
    for (int i = 0; i < 8; ++i) z[i] = (_Float16)0;
    v8h* p1 = (v8h*)spk1d;                 // first 256 v8h = buffer 0
    v8h* p2 = (v8h*)spk2d;
#pragma unroll
    for (int j = 0; j < 4; ++j) { p1[gt + 64 * j] = z; p2[gt + 64 * j] = z; }
  }
  __syncthreads();

  // ---- scalars (clip alpha/beta to [0,1] like reference) ----
  const float a1f = fminf(fmaxf(alpha1[0], 0.f), 1.f);
  const float b1f = fminf(fmaxf(beta1[0], 0.f), 1.f);
  const float t1  = thr1p[0];
  const float a2f = fminf(fmaxf(alpha2[0], 0.f), 1.f);
  const float b2f = fminf(fmaxf(beta2[0], 0.f), 1.f);
  const float t2  = thr2p[0];

  const int col = lane & 15;
  const int hf  = lane >> 4;
  const int nb0 = hlf * 64;          // first hidden column owned by this wave

  // ---- cur1_in = state @ w_fc1.T for our 4 tiles (K=6 plain FMAs) ----
  v8f cur1[4];
  {
    float st[8][SDIM];
#pragma unroll
    for (int i = 0; i < 8; ++i)
#pragma unroll
      for (int k = 0; k < SDIM; ++k) st[i][k] = stbuf[(i + hf * 8) * 8 + k];
#pragma unroll
    for (int n = 0; n < 4; ++n) {
      const float* wr = wfc1s + (nb0 + n * 16 + col) * 8;
      float w0 = wr[0], w1 = wr[1], w2 = wr[2], w3 = wr[3], w4 = wr[4], w5 = wr[5];
#pragma unroll
      for (int i = 0; i < 8; ++i) {
        cur1[n][i] = st[i][0] * w0 + st[i][1] * w1 + st[i][2] * w2 +
                     st[i][3] * w3 + st[i][4] * w4 + st[i][5] * w5;
      }
    }
  }

  // ---- per-column head weights for our 4 tiles ----
  float wm[4], wsd[4];
#pragma unroll
  for (int n = 0; n < 4; ++n) {
    wm[n]  = wmeans[nb0 + n * 16 + col];
    wsd[n] = wstds[nb0 + n * 16 + col];
  }

  // ---- recurrent state (WMMA C layout), 16 rows x 64 cols per wave ----
  v8f syn1[4], m1[4], syn2[4], m2[4];
  float pm[8], ps[8];
#pragma unroll
  for (int n = 0; n < 4; ++n)
#pragma unroll
    for (int i = 0; i < 8; ++i) { syn1[n][i] = 0.f; m1[n][i] = 0.f; syn2[n][i] = 0.f; m2[n][i] = 0.f; }
#pragma unroll
  for (int i = 0; i < 8; ++i) { pm[i] = 0.f; ps[i] = 0.f; }

  // ================== time-step loop ==================
  for (int t = 0; t < TSTEPS; ++t) {
    const int rb = t & 1;            // read buffer index
    const int wb = rb ^ 1;           // write buffer index
    const _Float16* s1r = spk1d + rb * 2048;
    _Float16*       s1w = spk1d + wb * 2048;
    const _Float16* s2r = spk2d + rb * 2048;
    _Float16*       s2w = spk2d + wb * 2048;

    // ---- layer 1: syn1 = a1*syn1 + cur1 + spk1_old @ w_rec1^T ----
#pragma unroll
    for (int n = 0; n < 4; ++n)
#pragma unroll
      for (int i = 0; i < 8; ++i) syn1[n][i] = a1f * syn1[n][i] + cur1[n][i];
#pragma unroll
    for (int kc = 0; kc < 4; ++kc) {       // K-outer: 1 A chunk live at a time
      v16h A = load_a(s1r, lane, kc * 32);
#pragma unroll
      for (int n = 0; n < 4; ++n)
        syn1[n] = wmma_f16(A, load_b(wrec1, lane, nb0 + n * 16, kc * 32), syn1[n]);
    }
    // mem1 update + spike -> LDS (this wave writes its 64 columns)
#pragma unroll
    for (int n = 0; n < 4; ++n) {
#pragma unroll
      for (int i = 0; i < 8; ++i) {
        float old = m1[n][i];
        float m   = b1f * old + syn1[n][i] - ((old - t1) > 0.f ? t1 : 0.f);
        m1[n][i]  = m;
        s1w[(i + hf * 8) * H + nb0 + n * 16 + col] =
            (_Float16)(((m - t1) > 0.f) ? 1.f : 0.f);
      }
    }
    __syncthreads();   // pair's spk1_new fully visible

    // ---- layer 2: syn2 = a2*syn2 + spk1_new @ w_fc2^T + spk2_old @ w_rec2^T ----
#pragma unroll
    for (int n = 0; n < 4; ++n)
#pragma unroll
      for (int i = 0; i < 8; ++i) syn2[n][i] = a2f * syn2[n][i];
#pragma unroll
    for (int kc = 0; kc < 4; ++kc) {
      v16h A = load_a(s1w, lane, kc * 32);
#pragma unroll
      for (int n = 0; n < 4; ++n)
        syn2[n] = wmma_f16(A, load_b(wfc2s, lane, nb0 + n * 16, kc * 32), syn2[n]);
    }
#pragma unroll
    for (int kc = 0; kc < 4; ++kc) {
      v16h A = load_a(s2r, lane, kc * 32);
#pragma unroll
      for (int n = 0; n < 4; ++n)
        syn2[n] = wmma_f16(A, load_b(wrec2, lane, nb0 + n * 16, kc * 32), syn2[n]);
    }
    // mem2 update, spike -> LDS, fold head dot-products
#pragma unroll
    for (int n = 0; n < 4; ++n) {
#pragma unroll
      for (int i = 0; i < 8; ++i) {
        float old = m2[n][i];
        float m   = b2f * old + syn2[n][i] - ((old - t2) > 0.f ? t2 : 0.f);
        m2[n][i]  = m;
        float sp  = ((m - t2) > 0.f) ? 1.f : 0.f;
        pm[i] += sp * wm[n];
        ps[i] += sp * wsd[n];
        s2w[(i + hf * 8) * H + nb0 + n * 16 + col] = (_Float16)sp;
      }
    }
    __syncthreads();   // end of step: reads of rb done, writes of wb visible
  }

  // ---- reduce partial dots across 16 lanes of each half-wave ----
#pragma unroll
  for (int msk = 1; msk < 16; msk <<= 1) {
#pragma unroll
    for (int i = 0; i < 8; ++i) {
      pm[i] += __shfl_xor(pm[i], msk, 32);
      ps[i] += __shfl_xor(ps[i], msk, 32);
    }
  }
  // combine the wave pair through group scratch (reuse stbuf: 32 floats)
  if (hlf == 1 && (lane & 15) == 0) {
#pragma unroll
    for (int i = 0; i < 8; ++i) {
      stbuf[hf * 8 + i]      = pm[i];
      stbuf[16 + hf * 8 + i] = ps[i];
    }
  }
  __syncthreads();
  if (hlf == 0 && (lane & 15) == 0) {
    const int mb = rowbase + hf * 8;   // rows mb..mb+7
#pragma unroll
    for (int i = 0; i < 8; ++i) {
      float tm  = pm[i] + stbuf[hf * 8 + i];
      float tsd = ps[i] + stbuf[16 + hf * 8 + i];
      out[mb + i]         = tanhf(tm * 0.125f);
      float x = tsd * 0.125f + 2.0f;
      out[BATCH + mb + i] = 1.9f / (1.0f + __expf(-x)) + 0.1f;
    }
  }
}

extern "C" void kernel_launch(void* const* d_in, const int* in_sizes, int n_in,
                              void* d_out, int out_size, void* d_ws, size_t ws_size,
                              hipStream_t stream) {
  const float* state  = (const float*)d_in[0];
  const float* w_fc1  = (const float*)d_in[1];
  const float* w_rec1 = (const float*)d_in[2];
  const float* w_fc2  = (const float*)d_in[3];
  const float* w_rec2 = (const float*)d_in[4];
  const float* w_mean = (const float*)d_in[5];
  const float* w_std  = (const float*)d_in[6];
  const float* alpha1 = (const float*)d_in[7];
  const float* beta1  = (const float*)d_in[8];
  const float* thr1   = (const float*)d_in[9];
  const float* alpha2 = (const float*)d_in[10];
  const float* beta2  = (const float*)d_in[11];
  const float* thr2   = (const float*)d_in[12];
  float* out = (float*)d_out;

  (void)in_sizes; (void)n_in; (void)out_size; (void)d_ws; (void)ws_size;

  // >64KB dynamic LDS (gfx1250 WGP has 320KB)
  hipFuncSetAttribute((const void*)snn_critic_kernel,
                      hipFuncAttributeMaxDynamicSharedMemorySize, LDS_TOTAL);

  dim3 grid(BATCH / 128), block(512);
  hipLaunchKernelGGL(snn_critic_kernel, grid, block, LDS_TOTAL, stream,
                     state, w_fc1, w_rec1, w_fc2, w_rec2, w_mean, w_std,
                     alpha1, beta1, thr1, alpha2, beta2, thr2, out);
}
